// FusedLinearAttentionBlock_50508815401129
// MI455X (gfx1250) — matrix operand order repaired and checked by reading, not compiled, not run on hardware
//
#include <hip/hip_runtime.h>
#include <hip/hip_bf16.h>
#include <cstdint>

// ---------------------------------------------------------------------------
// FusedLinearAttentionBlock for MI455X (gfx1250), bf16 WMMA pipeline.
// B=4, S=4096, D=1024, H=16, DH=64.
// TDM (tensor_load_to_lds) A-tile staging, double-buffered so the DMA of
// chunk k+1 overlaps the WMMA compute of chunk k (one barrier per chunk).
// ---------------------------------------------------------------------------

typedef __attribute__((ext_vector_type(16))) __bf16 v16bf;
typedef __attribute__((ext_vector_type(8)))  float  v8f;
typedef __attribute__((ext_vector_type(4)))  unsigned int v4u;
typedef __attribute__((ext_vector_type(8)))  int v8i;
typedef __attribute__((ext_vector_type(4)))  int v4i;

typedef unsigned short u16_t;
typedef unsigned int   u32_t;

#define BQ   4
#define SEQ  4096
#define DIM  1024
#define NH   16
#define DHE  64
#define BH   (BQ * NH)     // 64
#define MTOT (BQ * SEQ)    // 16384

#ifndef __has_builtin
#define __has_builtin(x) 0
#endif

#if __has_builtin(__builtin_amdgcn_tensor_load_to_lds) && \
    __has_builtin(__builtin_amdgcn_s_wait_tensorcnt)
#define HAVE_TDM 1
#else
#define HAVE_TDM 0
#endif

#if __has_builtin(__builtin_amdgcn_global_load_async_to_lds_b128)
#define HAVE_ASYNC 1
#else
#define HAVE_ASYNC 0
#endif

__device__ __forceinline__ u16_t f32_to_bf16(float f) {
  u32_t u = __builtin_bit_cast(u32_t, f);
  u32_t r = u + 0x7FFFu + ((u >> 16) & 1u);   // round-to-nearest-even
  return (u16_t)(r >> 16);
}

union FragU {
  uint4 u[2];
  v16bf f;
};

// A-matrix fragment (16x32 bf16, MxK). Lanes 0-15: K 0-7 & 16-23;
// lanes 16-31: K 8-15 & 24-31. Two contiguous 16B loads per lane.
__device__ __forceinline__ v16bf load_frag_a(const u16_t* base, int rstride, int lane) {
  const int r  = lane & 15;
  const int k0 = (lane & 16) ? 8 : 0;
  const u16_t* p = base + r * rstride + k0;
  FragU u;
  u.u[0] = *(const uint4*)(p);
  u.u[1] = *(const uint4*)(p + 16);
  return u.f;
}

// B-matrix fragment (32x16 bf16, KxN), tile stored transposed as [n][k].
// Lanes 0-15: K 0-7 & 16-23; lanes 16-31: K 8-15 & 24-31. Two contiguous 16B loads.
__device__ __forceinline__ v16bf load_frag_b(const u16_t* base, int rstride, int lane) {
  const int r  = lane & 15;
  const int k0 = (lane & 16) ? 8 : 0;
  const u16_t* p = base + r * rstride + k0;
  FragU u;
  u.u[0] = *(const uint4*)(p);
  u.u[1] = *(const uint4*)(p + 16);
  return u.f;
}

__device__ __forceinline__ u32_t lds_offset_of(const void* p) {
  return (u32_t)(uintptr_t)(__attribute__((address_space(3))) const void*)p;
}

#if HAVE_TDM
// Issue one TDM 2D tile load: rows x cols (bf16 elements) from a row-major
// tensor with pitch row_pitch_elems, packed into LDS at lds_off.
// D# layout per CDNA5 ISA ch. 8 (group0: count/lds/global/type,
// group1: data_size=2B, dims, tile dims, stride).
__device__ __forceinline__ void tdm_load_tile_2d(const void* gsrc, u32_t lds_off,
                                                 int rows, int cols,
                                                 int row_pitch_elems,
                                                 int tensor_rows) {
  unsigned long long ga = (unsigned long long)(uintptr_t)gsrc;
  v4u g0;
  g0.x = 1u;                                      // count=1, user descriptor
  g0.y = lds_off;                                 // lds_addr (bytes)
  g0.z = (u32_t)(ga & 0xFFFFFFFFu);               // global_addr[31:0]
  g0.w = (u32_t)((ga >> 32) & 0x01FFFFFFu) | (2u << 30);  // addr[56:32], type=2
  const u32_t td0 = (u32_t)row_pitch_elems;       // tensor_dim0
  const u32_t td1 = (u32_t)tensor_rows;           // tensor_dim1
  v8i g1;
  g1[0] = (int)(1u << 16);                        // data_size=1 -> 2 bytes
  g1[1] = (int)((td0 & 0xFFFFu) << 16);           // tensor_dim0[15:0]
  g1[2] = (int)((td0 >> 16) | ((td1 & 0xFFFFu) << 16));   // dim0 hi, dim1 lo
  g1[3] = (int)((td1 >> 16) | ((u32_t)cols << 16));       // dim1 hi, tile_dim0
  g1[4] = (int)((u32_t)rows & 0xFFFFu);           // tile_dim1 (tile_dim2=0)
  g1[5] = (int)row_pitch_elems;                   // tensor_dim0_stride[31:0]
  g1[6] = 0;
  g1[7] = 0;
  const v4i z4 = {0, 0, 0, 0};
#if __has_include(<hip/amd_detail/amd_gfx1250_TDM.h>)
  const v8i z8 = {0, 0, 0, 0, 0, 0, 0, 0};
  __builtin_amdgcn_tensor_load_to_lds(g0, g1, z4, z4, z8, 0);   // 6-arg toolchain
#else
  __builtin_amdgcn_tensor_load_to_lds(g0, g1, z4, z4, 0);       // 5-arg toolchain
#endif
}
#endif

// Wait for the in-flight A-tile staging issued by this wave (path-dependent).
__device__ __forceinline__ void wait_stage_a(int w) {
#if HAVE_TDM
  if (w == 0) __builtin_amdgcn_s_wait_tensorcnt(0);
#elif HAVE_ASYNC
  (void)w;
#if __has_builtin(__builtin_amdgcn_s_wait_asynccnt)
  __builtin_amdgcn_s_wait_asynccnt(0);
#else
  asm volatile("s_wait_asynccnt 0x0" ::: "memory");
#endif
#else
  (void)w;
#endif
}

// ---------------------------------------------------------------------------
// Elementwise converts
// ---------------------------------------------------------------------------
__global__ __launch_bounds__(256) void cvt_f32_bf16(const float* __restrict__ src,
                                                    u16_t* __restrict__ dst, int n) {
  int i = blockIdx.x * 256 + threadIdx.x;
  if (i < n) dst[i] = f32_to_bf16(src[i]);
}

// WoT[k*D + n] = Wo[n*D + k]   (out proj is attn @ Wo^T)
__global__ __launch_bounds__(256) void transpose_cvt(const float* __restrict__ Wo,
                                                     u16_t* __restrict__ WoT) {
  int i = blockIdx.x * 256 + threadIdx.x;
  if (i < DIM * DIM) {
    int k = i >> 10;
    int n = i & 1023;
    WoT[i] = f32_to_bf16(Wo[(size_t)n * DIM + k]);
  }
}

__global__ __launch_bounds__(256) void zero_f32(float* __restrict__ p, int n) {
  int i = blockIdx.x * 256 + threadIdx.x;
  if (i < n) p[i] = 0.0f;
}

// ---------------------------------------------------------------------------
// GEMM: C[16384,1024] = A[16384,1024] * B[1024,1024], bf16 in, bf16/f32 out.
// BM=BN=128, BK=32. 256 threads = 8 waves (4x2), wave tile 32x64 (2x4 WMMA).
// Double-buffered: TDM A-load + B transpose-stage of chunk k+1 overlap the
// WMMA compute of chunk k; exactly one workgroup barrier per chunk.
// ---------------------------------------------------------------------------
template <bool OUT_F32>
__global__ __launch_bounds__(256) void gemm_bf16(const u16_t* __restrict__ A,
                                                 const u16_t* __restrict__ B,
                                                 void* __restrict__ Cout) {
  constexpr int K = 1024;
  constexpr int N = 1024;
  __shared__ u16_t Als[2][128 * 32];   // A tile, row-major [m][k]
  __shared__ u16_t Bls[2][128 * 32];   // B tile, transposed [n][k]

  const int tid  = threadIdx.x;
  const int lane = tid & 31;
  const int w    = tid >> 5;
  const int wm   = w & 3;
  const int wn   = w >> 2;
  const int bm   = blockIdx.x * 128;
  const int bn   = blockIdx.y * 128;

  // ---- staging helpers ----------------------------------------------------
  auto stageA = [&](int kc, int buf) {
#if HAVE_TDM
    if (w == 0) {
      tdm_load_tile_2d(A + (size_t)bm * K + kc, lds_offset_of(&Als[buf][0]),
                       /*rows=*/128, /*cols=*/32, /*pitch=*/K, /*trows=*/MTOT);
    }
#elif HAVE_ASYNC
    int u = tid;
#pragma unroll
    for (int it = 0; it < 2; ++it) {
      int row = u >> 2;
      int q   = u & 3;
      __builtin_amdgcn_global_load_async_to_lds_b128(
          (__attribute__((address_space(1))) void*)(void*)(
              const_cast<u16_t*>(A) + (size_t)(bm + row) * K + kc + q * 8),
          (__attribute__((address_space(3))) void*)(&Als[buf][row * 32 + q * 8]),
          0, 0);
      u += 256;
    }
#else
    int u = tid;
#pragma unroll
    for (int it = 0; it < 2; ++it) {
      int row = u >> 2;
      int q   = u & 3;
      const uint4* g = (const uint4*)(A + (size_t)(bm + row) * K + kc + q * 8);
      *(uint4*)(&Als[buf][row * 32 + q * 8]) = g[0];
      u += 256;
    }
#endif
  };
  auto stageB = [&](int kc, int buf) {
    int krow = tid >> 3;          // 0..31
    int cb   = (tid & 7) * 16;    // 0..112
    const uint4* g = (const uint4*)(B + (size_t)(kc + krow) * N + bn + cb);
    uint4 d0 = g[0];
    uint4 d1 = g[1];
    u16_t tmp[16];
    *(uint4*)&tmp[0] = d0;
    *(uint4*)&tmp[8] = d1;
#pragma unroll
    for (int i = 0; i < 16; ++i) Bls[buf][(cb + i) * 32 + krow] = tmp[i];
  };

  const v8f vzero = {};
  v8f acc[2][4];
#pragma unroll
  for (int mi = 0; mi < 2; ++mi)
#pragma unroll
    for (int ni = 0; ni < 4; ++ni) acc[mi][ni] = vzero;

  // prologue: stage chunk 0 into buffer 0
  stageA(0, 0);
  stageB(0, 0);

  for (int kc = 0; kc < K; kc += 32) {
    const int cur = (kc >> 5) & 1;
    // complete in-flight staging of chunk `cur`, make LDS visible to all waves
    wait_stage_a(w);
    __syncthreads();   // also waits DScnt (covers B-stage stores)

    // kick off next chunk's staging into the other buffer (overlaps compute)
    if (kc + 32 < K) {
      stageA(kc + 32, cur ^ 1);
      stageB(kc + 32, cur ^ 1);
      __builtin_prefetch(B + (size_t)(kc + 64 + (tid >> 3)) * N + bn, 0, 1);
    }

    v16bf af[2], bfr[4];
#pragma unroll
    for (int mi = 0; mi < 2; ++mi)
      af[mi] = load_frag_a(&Als[cur][(wm * 32 + mi * 16) * 32], 32, lane);
#pragma unroll
    for (int ni = 0; ni < 4; ++ni)
      bfr[ni] = load_frag_b(&Bls[cur][(wn * 64 + ni * 16) * 32], 32, lane);

#pragma unroll
    for (int mi = 0; mi < 2; ++mi)
#pragma unroll
      for (int ni = 0; ni < 4; ++ni)
        acc[mi][ni] = __builtin_amdgcn_wmma_f32_16x16x32_bf16(
            false, af[mi], false, bfr[ni], (short)0, acc[mi][ni], false, false);
  }

  // ---- epilogue: C layout VGPR i -> M = i + (lane<16 ? 0 : 8), N = lane&15
  const int cl = lane & 15;
  const int rh = (lane & 16) ? 8 : 0;
#pragma unroll
  for (int mi = 0; mi < 2; ++mi)
#pragma unroll
    for (int ni = 0; ni < 4; ++ni) {
      int row = bm + wm * 32 + mi * 16 + rh;
      int col = bn + wn * 64 + ni * 16 + cl;
#pragma unroll
      for (int i = 0; i < 8; ++i) {
        if constexpr (OUT_F32)
          ((float*)Cout)[(size_t)(row + i) * N + col] = acc[mi][ni][i];
        else
          ((u16_t*)Cout)[(size_t)(row + i) * N + col] = f32_to_bf16(acc[mi][ni][i]);
      }
    }
}

// ---------------------------------------------------------------------------
// kv partial: kv[b,h,d,e] += sum_{s in block} K[b,s,h,d] * V[b,s,h,e]
// grid = (BH, S/512), 128 threads = 4 waves; wave w owns d-strip w*16..+15.
// Double-buffered manual transpose staging (TDM cannot transpose 16-bit).
// ---------------------------------------------------------------------------
__global__ __launch_bounds__(128) void kv_partial(const u16_t* __restrict__ Kb,
                                                  const u16_t* __restrict__ Vb,
                                                  float* __restrict__ kvf) {
  __shared__ u16_t Kls[2][64 * 32];   // [d][s]
  __shared__ u16_t Vls[2][64 * 32];   // [e][s]
  const int bh = blockIdx.x;
  const int b  = bh >> 4;
  const int h  = bh & 15;
  const int s0 = blockIdx.y * 512;
  const int tid  = threadIdx.x;
  const int lane = tid & 31;
  const int w    = tid >> 5;
  const size_t head = (size_t)b * SEQ * DIM + (size_t)h * DHE;

  auto stage = [&](int sc, int buf) {
    const int srow = tid >> 2;         // 0..31
    const int cb   = (tid & 3) * 16;   // 0,16,32,48
    const size_t gidx = head + (size_t)(s0 + sc + srow) * DIM + cb;
    {
      const uint4* gk = (const uint4*)(Kb + gidx);
      uint4 d0 = gk[0], d1 = gk[1];
      u16_t tmp[16];
      *(uint4*)&tmp[0] = d0;
      *(uint4*)&tmp[8] = d1;
#pragma unroll
      for (int i = 0; i < 16; ++i) Kls[buf][(cb + i) * 32 + srow] = tmp[i];
    }
    {
      const uint4* gv = (const uint4*)(Vb + gidx);
      uint4 d0 = gv[0], d1 = gv[1];
      u16_t tmp[16];
      *(uint4*)&tmp[0] = d0;
      *(uint4*)&tmp[8] = d1;
#pragma unroll
      for (int i = 0; i < 16; ++i) Vls[buf][(cb + i) * 32 + srow] = tmp[i];
    }
  };

  const v8f vzero = {};
  v8f acc[4];
#pragma unroll
  for (int ni = 0; ni < 4; ++ni) acc[ni] = vzero;

  stage(0, 0);
  for (int sc = 0; sc < 512; sc += 32) {
    const int cur = (sc >> 5) & 1;
    __syncthreads();   // DScnt wait + barrier: staging of `cur` now visible
    if (sc + 32 < 512) stage(sc + 32, cur ^ 1);

    v16bf af = load_frag_a(&Kls[cur][(w * 16) * 32], 32, lane);  // A = K^T
#pragma unroll
    for (int ni = 0; ni < 4; ++ni) {
      v16bf bfr = load_frag_b(&Vls[cur][(ni * 16) * 32], 32, lane);  // B = V
      acc[ni] = __builtin_amdgcn_wmma_f32_16x16x32_bf16(
          false, af, false, bfr, (short)0, acc[ni], false, false);
    }
  }

  // cross-s-block reduction via global float atomics
  const int cl = lane & 15;
  const int rh = (lane & 16) ? 8 : 0;
  float* dst = kvf + (size_t)bh * 4096;
#pragma unroll
  for (int ni = 0; ni < 4; ++ni) {
#pragma unroll
    for (int i = 0; i < 8; ++i) {
      int d = w * 16 + rh + i;
      int e = ni * 16 + cl;
      atomicAdd(&dst[d * 64 + e], acc[ni][i]);
    }
  }
}

// kvT[bh][e][d] = bf16( kvf[bh][d][e] * 1/sqrt(64) )
__global__ __launch_bounds__(256) void kv_finalize(const float* __restrict__ kvf,
                                                   u16_t* __restrict__ kvT) {
  int idx = blockIdx.x * 256 + threadIdx.x;
  if (idx < BH * DHE * DHE) {
    int bh = idx >> 12;
    int e  = (idx >> 6) & 63;
    int d  = idx & 63;
    kvT[idx] = f32_to_bf16(kvf[(bh << 12) + (d << 6) + e] * 0.125f);
  }
}

// ---------------------------------------------------------------------------
// attn[b,s,h*64+e] = sum_d Q[b,s,h,d] * kv[b,h,d,e]   (kv pre-scaled)
// grid = (S/128, BH), 256 threads = 8 waves, wave owns a 16-row strip.
// Fragments straight from global (kvT stored transposed -> contiguous).
// ---------------------------------------------------------------------------
__global__ __launch_bounds__(256) void q_kv(const u16_t* __restrict__ Qb,
                                            const u16_t* __restrict__ kvT,
                                            u16_t* __restrict__ attn) {
  const int bh = blockIdx.y;
  const int b  = bh >> 4;
  const int h  = bh & 15;
  const int tid  = threadIdx.x;
  const int lane = tid & 31;
  const int w    = tid >> 5;
  const int mrow = blockIdx.x * 128 + w * 16;

  const u16_t* Qp = Qb + (size_t)b * SEQ * DIM + (size_t)h * DHE;
  const u16_t* Bp = kvT + (size_t)bh * 4096;     // [e][d]

  const v8f vzero = {};
  v8f acc[4];
#pragma unroll
  for (int ni = 0; ni < 4; ++ni) acc[ni] = vzero;

#pragma unroll
  for (int kc = 0; kc < 64; kc += 32) {
    v16bf af = load_frag_a(Qp + (size_t)mrow * DIM + kc, DIM, lane);
#pragma unroll
    for (int ni = 0; ni < 4; ++ni) {
      v16bf bfr = load_frag_b(Bp + (ni * 16) * 64 + kc, 64, lane);
      acc[ni] = __builtin_amdgcn_wmma_f32_16x16x32_bf16(
          false, af, false, bfr, (short)0, acc[ni], false, false);
    }
  }

  const int cl = lane & 15;
  const int rh = (lane & 16) ? 8 : 0;
#pragma unroll
  for (int ni = 0; ni < 4; ++ni) {
#pragma unroll
    for (int i = 0; i < 8; ++i) {
      size_t row = (size_t)b * SEQ + mrow + rh + i;
      attn[row * DIM + h * DHE + ni * 16 + cl] = f32_to_bf16(acc[ni][i]);
    }
  }
}

// ---------------------------------------------------------------------------
// Host-side orchestration
// ---------------------------------------------------------------------------
extern "C" void kernel_launch(void* const* d_in, const int* in_sizes, int n_in,
                              void* d_out, int out_size, void* d_ws, size_t ws_size,
                              hipStream_t stream) {
  (void)in_sizes; (void)n_in; (void)out_size; (void)ws_size;
  const float* x  = (const float*)d_in[0];
  const float* Wq = (const float*)d_in[1];
  const float* Wk = (const float*)d_in[2];
  const float* Wv = (const float*)d_in[3];
  const float* Wo = (const float*)d_in[4];

  char* ws = (char*)d_ws;
  // workspace layout (bytes)
  u16_t* xb   = (u16_t*)(ws + 0);            // 32 MB  (reused later as attn)
  u16_t* wqb  = (u16_t*)(ws + 33554432);     //  2 MB
  u16_t* wkb  = (u16_t*)(ws + 35651584);     //  2 MB
  u16_t* wvb  = (u16_t*)(ws + 37748736);     //  2 MB
  u16_t* wotb = (u16_t*)(ws + 39845888);     //  2 MB
  u16_t* qb   = (u16_t*)(ws + 41943040);     // 32 MB
  u16_t* kb   = (u16_t*)(ws + 75497472);     // 32 MB
  u16_t* vb   = (u16_t*)(ws + 109051904);    // 32 MB
  float* kvf  = (float*)(ws + 142606336);    //  1 MB
  u16_t* kvt  = (u16_t*)(ws + 143654912);    // 512 KB
  u16_t* attn = xb;                           // reuse: x_bf16 dead after QKV

  const int NX = MTOT * DIM;      // 16,777,216
  const int NW = DIM * DIM;       //  1,048,576

  // 1) fp32 -> bf16 converts
  cvt_f32_bf16<<<(NX + 255) / 256, 256, 0, stream>>>(x,  xb,  NX);
  cvt_f32_bf16<<<(NW + 255) / 256, 256, 0, stream>>>(Wq, wqb, NW);
  cvt_f32_bf16<<<(NW + 255) / 256, 256, 0, stream>>>(Wk, wkb, NW);
  cvt_f32_bf16<<<(NW + 255) / 256, 256, 0, stream>>>(Wv, wvb, NW);
  transpose_cvt<<<(NW + 255) / 256, 256, 0, stream>>>(Wo, wotb);

  // 2) QKV projections (bf16 WMMA GEMMs, TDM double-buffered)
  dim3 gg(MTOT / 128, DIM / 128);
  gemm_bf16<false><<<gg, 256, 0, stream>>>(xb, wqb, qb);
  gemm_bf16<false><<<gg, 256, 0, stream>>>(xb, wkb, kb);
  gemm_bf16<false><<<gg, 256, 0, stream>>>(xb, wvb, vb);

  // 3) kv = K^T V  (zero, partial-accumulate with atomics, finalize+scale+T)
  zero_f32<<<(BH * DHE * DHE + 255) / 256, 256, 0, stream>>>(kvf, BH * DHE * DHE);
  kv_partial<<<dim3(BH, SEQ / 512), 128, 0, stream>>>(kb, vb, kvf);
  kv_finalize<<<(BH * DHE * DHE + 255) / 256, 256, 0, stream>>>(kvf, kvt);

  // 4) out_head = Q @ kv  -> attn buffer [B,S,D] bf16
  q_kv<<<dim3(SEQ / 128, BH), 256, 0, stream>>>(qb, kvt, attn);

  // 5) final projection: d_out = attn @ Wo^T (fp32 epilogue)
  gemm_bf16<true><<<gg, 256, 0, stream>>>(attn, wotb, (float*)d_out);
}
